// TFFMBlock3D_10823317586770
// MI455X (gfx1250) — matrix operand order repaired and checked by reading, not compiled, not run on hardware
//
#include <hip/hip_runtime.h>
#include <hip/hip_bf16.h>
#include <math.h>

// ---------------------------------------------------------------------------
// TFFMBlock3D for MI455X (gfx1250, wave32, WMMA).
//
// Pipeline:
//   k_pool      : xp[b,c,node] = mean over 8^3 block of x        (134MB read)
//   k_compress  : nodes = W_c * xp + b_c ; unit-normalized f16 copy
//   k_sims      : S = nrm . nrm^T via v_wmma_f32_16x16x32_f16
//   k_topk      : per-row top-8 (deterministic tie: lowest j) -> bitmask
//   k_sym       : adj = A | A^T
//   k_gat_pre   : H = X W^T, e_l, e_r
//   k_gat_attn  : masked leaky-relu softmax attention over neighbor bits
//   k_inorm     : instance norm over 512 nodes (+gelu / +residual-add modes)
//   k_bfrag     : pack expand/gate weights into WMMA-B f16 fragments
//   k_fused_out : trilinear upsample -> expand(WMMA) -> gate(WMMA) -> sigmoid
//                 -> residual, 12 WMMAs per 16-voxel tile    (134MB r + 134MB w)
// ---------------------------------------------------------------------------

typedef __attribute__((ext_vector_type(16))) _Float16 v16h;
typedef __attribute__((ext_vector_type(8)))  float    v8f;

#define B_     2
#define C_     64
#define HID    32
#define D_     64
#define G_     8
#define N_     512
#define KNN    8
#define HEADS_ 4
#define HD     8
#define EPS_   1e-5f

union FragAB { v16h v; unsigned u[8]; uint4 q[2]; };

static __device__ __forceinline__ int laneId() { return threadIdx.x & 31; }
static __device__ __forceinline__ int waveId() { return threadIdx.x >> 5; }

static __device__ __forceinline__ float wsum(float v){
#pragma unroll
  for (int m = 16; m >= 1; m >>= 1) v += __shfl_xor(v, m, 32);
  return v;
}
static __device__ __forceinline__ float wmaxr(float v){
#pragma unroll
  for (int m = 16; m >= 1; m >>= 1) v = fmaxf(v, __shfl_xor(v, m, 32));
  return v;
}

// ---------------------------------------------------------------------------
// 1) Pool x over 8x8x8 blocks: xp[b][c][node], node = gd*64+gh*8+gw.
//    One wave per (b,c,node). 65536 waves total.
// ---------------------------------------------------------------------------
__global__ __launch_bounds__(256) void k_pool(const float* __restrict__ x,
                                              float* __restrict__ xp){
  int gid = blockIdx.x * 8 + waveId();          // 0..65535
  int b = gid >> 15;
  int c = (gid >> 9) & 63;
  int node = gid & 511;
  int gd = node >> 6, gh = (node >> 3) & 7, gw = node & 7;
  const float* base = x + (((size_t)(b * 64 + c) * 64 + gd * 8) * 4096)
                        + gh * 8 * 64 + gw * 8;
  int l = laneId();
  float s = 0.f;
#pragma unroll
  for (int it = 0; it < 16; ++it){
    int idx = l + it * 32;                      // 0..511 within the block
    int dd = idx >> 6, hh = (idx >> 3) & 7, ww = idx & 7;
    s += base[dd * 4096 + hh * 64 + ww];
  }
  s = wsum(s);
  if (l == 0) xp[gid] = s * (1.0f / 512.0f);
}

// ---------------------------------------------------------------------------
// 2) Compress (conv1x1 commuted past the pooling) + cosine-normalized f16 copy.
//    One thread per (b,n).
// ---------------------------------------------------------------------------
__global__ __launch_bounds__(256) void k_compress(const float* __restrict__ xp,
                                                  const float* __restrict__ W,
                                                  const float* __restrict__ bias,
                                                  float* __restrict__ nodes,
                                                  _Float16* __restrict__ nrm16){
  int t = blockIdx.x * blockDim.x + threadIdx.x;
  if (t >= B_ * N_) return;
  int b = t >> 9, n = t & 511;
  float acc[HID];
#pragma unroll
  for (int o = 0; o < HID; ++o) acc[o] = bias[o];
  for (int c = 0; c < C_; ++c){
    float v = xp[(b * C_ + c) * N_ + n];
#pragma unroll
    for (int o = 0; o < HID; ++o) acc[o] += v * W[o * C_ + c];
  }
  float ss = 0.f;
#pragma unroll
  for (int o = 0; o < HID; ++o){ nodes[(size_t)t * HID + o] = acc[o]; ss += acc[o] * acc[o]; }
  float inv = 1.0f / fmaxf(sqrtf(ss), 1e-12f);
#pragma unroll
  for (int o = 0; o < HID; ++o) nrm16[(size_t)t * HID + o] = (_Float16)(acc[o] * inv);
}

// ---------------------------------------------------------------------------
// 3) Cosine similarity: one v_wmma_f32_16x16x32_f16 per 16x16 tile.
//    A (16x32): lane m=l&15, VGPR v -> K pair 2v + (l<16?0:16)
//    B (32x16): col n=l&15, VGPR v -> K pair 2v + (l<16?0:16)
// ---------------------------------------------------------------------------
__global__ __launch_bounds__(128) void k_sims(const _Float16* __restrict__ nrm,
                                              float* __restrict__ S){
  int tid = blockIdx.x * 4 + waveId();          // 0..2047
  int b  = tid >> 10;
  int it = (tid >> 5) & 31;
  int jt = tid & 31;
  int l = laneId();
  int m = l & 15;

  FragAB a, bb;
  const _Float16* arow = nrm + ((size_t)b * N_ + it * 16 + m) * HID;
#pragma unroll
  for (int v = 0; v < 8; ++v){
    int k0 = 2 * v + (l < 16 ? 0 : 16);
    a.u[v] = *(const unsigned*)(arow + k0);
  }
  const _Float16* brow = nrm + ((size_t)b * N_ + jt * 16 + m) * HID + (l < 16 ? 0 : 16);
#pragma unroll
  for (int v = 0; v < 8; ++v) bb.u[v] = *(const unsigned*)(brow + 2 * v);

  v8f cz = {0.f,0.f,0.f,0.f,0.f,0.f,0.f,0.f};
  v8f acc = __builtin_amdgcn_wmma_f32_16x16x32_f16(false, a.v, false, bb.v,
                                                   (short)0, cz, false, false);
  int j = jt * 16 + m;
#pragma unroll
  for (int r = 0; r < 8; ++r){
    int i = it * 16 + r + (l < 16 ? 0 : 8);
    S[((size_t)b * N_ + i) * N_ + j] = acc[r];
  }
}

// ---------------------------------------------------------------------------
// 4) Top-8 per row -> one-hot bitmask A (16 u32 words / row). One wave per row.
// ---------------------------------------------------------------------------
__global__ __launch_bounds__(256) void k_topk(const float* __restrict__ S,
                                              unsigned* __restrict__ A){
  int row = blockIdx.x * 8 + waveId();          // 0..1023
  int l = laneId();
  const float* s = S + (size_t)row * N_ + l * 16;
  float v[16];
#pragma unroll
  for (int k = 0; k < 16; ++k) v[k] = s[k];
  unsigned used = 0, myword = 0;
  for (int iter = 0; iter < KNN; ++iter){
    float bv = -3.0e38f; int bj = 1 << 30;
#pragma unroll
    for (int k = 0; k < 16; ++k){
      if (!((used >> k) & 1)){
        int j = l * 16 + k;
        if (v[k] > bv || (v[k] == bv && j < bj)){ bv = v[k]; bj = j; }
      }
    }
#pragma unroll
    for (int mm = 16; mm >= 1; mm >>= 1){
      float ov = __shfl_xor(bv, mm, 32);
      int   oj = __shfl_xor(bj, mm, 32);
      if (ov > bv || (ov == bv && oj < bj)){ bv = ov; bj = oj; }
    }
    if ((bj >> 4) == l) used |= 1u << (bj & 15);
    if ((bj >> 5) == l) myword |= 1u << (bj & 31);
  }
  if (l < 16) A[row * 16 + l] = myword;
}

// ---------------------------------------------------------------------------
// 5) Symmetrize: adj = A | A^T. One thread per row.
// ---------------------------------------------------------------------------
__global__ __launch_bounds__(256) void k_sym(const unsigned* __restrict__ A,
                                             unsigned* __restrict__ adj){
  int row = blockIdx.x * blockDim.x + threadIdx.x;
  if (row >= B_ * N_) return;
  int b = row >> 9, i = row & 511;
  int iw = i >> 5; unsigned ib = 1u << (i & 31);
  for (int w = 0; w < 16; ++w){
    unsigned mword = A[row * 16 + w];
    for (int jj = 0; jj < 32; ++jj){
      int j = w * 32 + jj;
      if (A[(b * N_ + j) * 16 + iw] & ib) mword |= 1u << jj;
    }
    adj[row * 16 + w] = mword;
  }
}

// ---------------------------------------------------------------------------
// 6) GAT projection: H = X W^T, e_l/e_r. One thread per (b,n).
// ---------------------------------------------------------------------------
__global__ __launch_bounds__(256) void k_gat_pre(const float* __restrict__ xin,
                                                 const float* __restrict__ W,
                                                 const float* __restrict__ al,
                                                 const float* __restrict__ ar,
                                                 float* __restrict__ H,
                                                 float* __restrict__ el,
                                                 float* __restrict__ er){
  int t = blockIdx.x * blockDim.x + threadIdx.x;
  if (t >= B_ * N_) return;
  float xv[HID];
#pragma unroll
  for (int k = 0; k < HID; ++k) xv[k] = xin[(size_t)t * HID + k];
  float h[HID];
#pragma unroll
  for (int o = 0; o < HID; ++o){
    float s = 0.f;
#pragma unroll
    for (int k = 0; k < HID; ++k) s += xv[k] * W[o * HID + k];
    h[o] = s; H[(size_t)t * HID + o] = s;
  }
#pragma unroll
  for (int hh = 0; hh < HEADS_; ++hh){
    float sl = 0.f, sr = 0.f;
#pragma unroll
    for (int d = 0; d < HD; ++d){
      sl += h[hh * HD + d] * al[hh * HD + d];
      sr += h[hh * HD + d] * ar[hh * HD + d];
    }
    el[t * HEADS_ + hh] = sl; er[t * HEADS_ + hh] = sr;
  }
}

// ---------------------------------------------------------------------------
// 7) GAT attention over neighbor bits. One wave per row i; lane l owns
//    j in [16l, 16l+16). Two passes: max, then exp-sum + weighted H gather.
// ---------------------------------------------------------------------------
__global__ __launch_bounds__(256) void k_gat_attn(const float* __restrict__ H,
                                                  const float* __restrict__ el,
                                                  const float* __restrict__ er,
                                                  const unsigned* __restrict__ adj,
                                                  float* __restrict__ out){
  int row = blockIdx.x * 8 + waveId();          // 0..1023
  int b = row >> 9;
  int l = laneId();
  unsigned word = adj[row * 16 + (l >> 1)];
  unsigned bits = (word >> ((l & 1) * 16)) & 0xFFFFu;

  float eli[HEADS_], emax[HEADS_];
#pragma unroll
  for (int h = 0; h < HEADS_; ++h){ eli[h] = el[row * HEADS_ + h]; emax[h] = -3.0e38f; }

  for (int k = 0; k < 16; ++k){
    if ((bits >> k) & 1){
      int j = l * 16 + k;
#pragma unroll
      for (int h = 0; h < HEADS_; ++h){
        float e = eli[h] + er[(b * N_ + j) * HEADS_ + h];
        e = e > 0.f ? e : 0.2f * e;
        emax[h] = fmaxf(emax[h], e);
      }
    }
  }
#pragma unroll
  for (int h = 0; h < HEADS_; ++h) emax[h] = wmaxr(emax[h]);

  float ssum[HEADS_], acc[HID];
#pragma unroll
  for (int h = 0; h < HEADS_; ++h) ssum[h] = 0.f;
#pragma unroll
  for (int c = 0; c < HID; ++c) acc[c] = 0.f;

  for (int k = 0; k < 16; ++k){
    if ((bits >> k) & 1){
      int j = l * 16 + k;
      const float* hj = H + (size_t)(b * N_ + j) * HID;
#pragma unroll
      for (int h = 0; h < HEADS_; ++h){
        float e = eli[h] + er[(b * N_ + j) * HEADS_ + h];
        e = e > 0.f ? e : 0.2f * e;
        float p = expf(e - emax[h]);
        ssum[h] += p;
#pragma unroll
        for (int d = 0; d < HD; ++d) acc[h * HD + d] += p * hj[h * HD + d];
      }
    }
  }
#pragma unroll
  for (int h = 0; h < HEADS_; ++h) ssum[h] = wsum(ssum[h]);
#pragma unroll
  for (int c = 0; c < HID; ++c) acc[c] = wsum(acc[c]);

  if (l == 0){
#pragma unroll
    for (int h = 0; h < HEADS_; ++h){
      float inv = 1.0f / ssum[h];
#pragma unroll
      for (int d = 0; d < HD; ++d)
        out[(size_t)row * HID + h * HD + d] = acc[h * HD + d] * inv;
    }
  }
}

// ---------------------------------------------------------------------------
// 8) Instance norm over 512 nodes per (b,c). gin is [b][n][c], vol is [b][c][n].
//    Optional: addSrc (vol += addSrc, for nodes_out) or geluDst ([b][n][c]).
// ---------------------------------------------------------------------------
__global__ __launch_bounds__(256) void k_inorm(const float* __restrict__ gin,
                                               float* __restrict__ vol,
                                               const float* __restrict__ addSrc,
                                               float* __restrict__ geluDst){
  int t = blockIdx.x * 8 + waveId();            // 0..63 : (b*HID + c)
  int b = t >> 5, c = t & 31;
  int l = laneId();
  float vals[16];
  float s = 0.f, ss = 0.f;
#pragma unroll
  for (int k = 0; k < 16; ++k){
    int n = l + 32 * k;
    float v = gin[((size_t)b * N_ + n) * HID + c];
    vals[k] = v; s += v; ss += v * v;
  }
  s = wsum(s); ss = wsum(ss);
  float mean = s * (1.f / 512.f);
  float var = ss * (1.f / 512.f) - mean * mean;
  float inv = rsqrtf(var + EPS_);
#pragma unroll
  for (int k = 0; k < 16; ++k){
    int n = l + 32 * k;
    float y = (vals[k] - mean) * inv;
    size_t vi = ((size_t)b * HID + c) * N_ + n;
    float o = y;
    if (addSrc) o = y + addSrc[vi];
    vol[vi] = o;
    if (geluDst){
      float g = 0.5f * y * (1.0f + erff(y * 0.70710678f));
      geluDst[((size_t)b * N_ + n) * HID + c] = g;
    }
  }
}

// ---------------------------------------------------------------------------
// 9) Pack a row-major [O][Kfull] f32 weight into WMMA-B f16 fragments.
//    dst dwords laid out [frag = t*nSlabs+s][lane][v], frag covers N-tile t,
//    K-slab s (32 rows). Each lane later loads its 32B chunk contiguously.
// ---------------------------------------------------------------------------
__global__ __launch_bounds__(256) void k_bfrag(const float* __restrict__ W,
                                               int O, int Kfull, int nSlabs,
                                               unsigned* __restrict__ dst){
  int id = blockIdx.x * blockDim.x + threadIdx.x;
  int total = (O / 16) * nSlabs * 256;
  if (id >= total) return;
  int v = id & 7;
  int lane = (id >> 3) & 31;
  int frag = id >> 8;
  int s = frag % nSlabs;
  int t = frag / nSlabs;
  int n = lane & 15;
  int o = t * 16 + n;
  int k0 = s * 32 + 2 * v + (lane < 16 ? 0 : 16);
  union { _Float16 h[2]; unsigned u; } p;
  p.h[0] = (_Float16)W[o * Kfull + k0];
  p.h[1] = (_Float16)W[o * Kfull + k0 + 1];
  dst[(size_t)(frag * 32 + lane) * 8 + v] = p.u;
}

// ---------------------------------------------------------------------------
// 10) Fused trilinear-upsample + expand(WMMA) + gate(WMMA) + sigmoid + residual.
//     Block = (b, d, h); 4 waves, each owns a 16-voxel w-tile.
//     12 v_wmma_f32_16x16x32_f16 per tile.
// ---------------------------------------------------------------------------
__global__ __launch_bounds__(128) void k_fused_out(
    const float* __restrict__ x, const float* __restrict__ nodesOut,
    const unsigned* __restrict__ bexp, const unsigned* __restrict__ bgate,
    const float* __restrict__ expand_b, const float* __restrict__ gate_b,
    float* __restrict__ out){
  __shared__ __attribute__((aligned(16))) float    nsl[2][2][HID][8];
  __shared__ __attribute__((aligned(16))) _Float16 A1[4][16][32];
  __shared__ __attribute__((aligned(16))) _Float16 A2[4][16][64];
  __shared__ __attribute__((aligned(16))) float    xt[4][64][16];

  int blk = blockIdx.x;
  int b = blk >> 12;
  int d = (blk >> 6) & 63;
  int h = blk & 63;

  float sd = fmaxf((d + 0.5f) * 0.125f - 0.5f, 0.f);
  int d0 = (int)sd; int d1 = min(d0 + 1, 7); float fd = sd - d0;
  float sh = fmaxf((h + 0.5f) * 0.125f - 0.5f, 0.f);
  int h0 = (int)sh; int h1 = min(h0 + 1, 7); float fh = sh - h0;

  for (int idx = threadIdx.x; idx < 2 * 2 * HID * 8; idx += 128){
    int gw = idx & 7;
    int c  = (idx >> 3) & 31;
    int hh = (idx >> 8) & 1;
    int dd = idx >> 9;
    int gd = dd ? d1 : d0;
    int gh = hh ? h1 : h0;
    nsl[dd][hh][c][gw] = nodesOut[((size_t)b * HID + c) * N_ + gd * 64 + gh * 8 + gw];
  }

  int wid = waveId(), l = laneId();
  int m = l & 15;
  int w0 = wid * 16;
  int wv = w0 + m;

  // residual tile: xt[o][m]  (64B-coalesced loads along w)
  const float* xbase = x + (((size_t)b * C_) * D_ + d) * 4096 + h * 64 + w0;
#pragma unroll
  for (int r = 0; r < 32; ++r){
    int idx = r * 32 + l;
    int o = idx >> 4, mm = idx & 15;
    xt[wid][o][mm] = xbase[(size_t)o * D_ * 4096 + mm];
  }
  __syncthreads();

  // trilinear -> A1 (f16, 16 voxels x 32 channels)
  float sw = fmaxf((wv + 0.5f) * 0.125f - 0.5f, 0.f);
  int g0 = (int)sw; int g1 = min(g0 + 1, 7); float fw = sw - g0;
  float wd0 = 1.f - fd, wh0 = 1.f - fh, ww0 = 1.f - fw;
  int cbase = (l < 16) ? 0 : 16;
#pragma unroll
  for (int cc = 0; cc < 16; ++cc){
    int c = cbase + cc;
    float v00 = nsl[0][0][c][g0] * ww0 + nsl[0][0][c][g1] * fw;
    float v01 = nsl[0][1][c][g0] * ww0 + nsl[0][1][c][g1] * fw;
    float v10 = nsl[1][0][c][g0] * ww0 + nsl[1][0][c][g1] * fw;
    float v11 = nsl[1][1][c][g0] * ww0 + nsl[1][1][c][g1] * fw;
    float v0 = v00 * wh0 + v01 * fh;
    float v1 = v10 * wh0 + v11 * fh;
    A1[wid][m][c] = (_Float16)(v0 * wd0 + v1 * fd);
  }
  __syncthreads();

  FragAB a1;
#pragma unroll
  for (int v = 0; v < 8; ++v){
    int k0 = 2 * v + (l < 16 ? 0 : 16);
    a1.u[v] = *(const unsigned*)&A1[wid][m][k0];
  }

  v8f cz = {0.f,0.f,0.f,0.f,0.f,0.f,0.f,0.f};
  v8f acc[4];
#pragma unroll
  for (int t = 0; t < 4; ++t){
    FragAB bf;
    const uint4* bp = (const uint4*)bexp + (size_t)(t * 32 + l) * 2;
    bf.q[0] = bp[0]; bf.q[1] = bp[1];
    acc[t] = __builtin_amdgcn_wmma_f32_16x16x32_f16(false, a1.v, false, bf.v,
                                                    (short)0, cz, false, false);
    float bias = expand_b[t * 16 + m];
#pragma unroll
    for (int r = 0; r < 8; ++r){
      float he = acc[t][r] + bias;
      acc[t][r] = he;
      A2[wid][r + (l < 16 ? 0 : 8)][t * 16 + m] = (_Float16)he;
    }
  }
  __syncthreads();

  FragAB a2lo, a2hi;
#pragma unroll
  for (int v = 0; v < 8; ++v){
    int k0 = 2 * v + (l < 16 ? 0 : 16);
    a2lo.u[v] = *(const unsigned*)&A2[wid][m][k0];
    a2hi.u[v] = *(const unsigned*)&A2[wid][m][k0 + 32];
  }

#pragma unroll
  for (int t = 0; t < 4; ++t){
    FragAB bg0, bg1;
    const uint4* bp0 = (const uint4*)bgate + (size_t)((t * 2 + 0) * 32 + l) * 2;
    const uint4* bp1 = (const uint4*)bgate + (size_t)((t * 2 + 1) * 32 + l) * 2;
    bg0.q[0] = bp0[0]; bg0.q[1] = bp0[1];
    bg1.q[0] = bp1[0]; bg1.q[1] = bp1[1];
    v8f c0 = __builtin_amdgcn_wmma_f32_16x16x32_f16(false, a2lo.v, false, bg0.v,
                                                    (short)0, cz, false, false);
    c0 = __builtin_amdgcn_wmma_f32_16x16x32_f16(false, a2hi.v, false, bg1.v,
                                                (short)0, c0, false, false);
    float gb = gate_b[t * 16 + m];
    int o = t * 16 + m;
#pragma unroll
    for (int r = 0; r < 8; ++r){
      int mr = r + (l < 16 ? 0 : 8);
      float he = acc[t][r];
      float gv = 1.0f / (1.0f + expf(-(c0[r] + gb)));
      xt[wid][o][mr] += gv * he;
    }
  }
  __syncthreads();

  float* obase = out + (((size_t)b * C_) * D_ + d) * 4096 + h * 64 + w0;
#pragma unroll
  for (int r = 0; r < 32; ++r){
    int idx = r * 32 + l;
    int o = idx >> 4, mm = idx & 15;
    obase[(size_t)o * D_ * 4096 + mm] = xt[wid][o][mm];
  }
}

// ---------------------------------------------------------------------------
extern "C" void kernel_launch(void* const* d_in, const int* in_sizes, int n_in,
                              void* d_out, int out_size, void* d_ws, size_t ws_size,
                              hipStream_t stream) {
  (void)in_sizes; (void)n_in; (void)out_size; (void)ws_size;
  const float* x          = (const float*)d_in[0];
  const float* compress_w = (const float*)d_in[1];
  const float* compress_b = (const float*)d_in[2];
  const float* gat1_W     = (const float*)d_in[3];
  const float* gat1_al    = (const float*)d_in[4];
  const float* gat1_ar    = (const float*)d_in[5];
  const float* gat2_W     = (const float*)d_in[6];
  const float* gat2_al    = (const float*)d_in[7];
  const float* gat2_ar    = (const float*)d_in[8];
  const float* expand_w   = (const float*)d_in[9];
  const float* expand_b   = (const float*)d_in[10];
  const float* gate_w     = (const float*)d_in[11];
  const float* gate_b     = (const float*)d_in[12];
  float* out = (float*)d_out;

  char* ws = (char*)d_ws;
  float*    xp       = (float*)(ws + 0);            // 256 KB
  float*    nodes    = (float*)(ws + 262144);       // 128 KB
  _Float16* nrm16    = (_Float16*)(ws + 393216);    //  64 KB
  float*    S        = (float*)(ws + 458752);       //   2 MB
  unsigned* A        = (unsigned*)(ws + 2555904);   //  64 KB
  unsigned* adj      = (unsigned*)(ws + 2621440);   //  64 KB
  float*    Hb       = (float*)(ws + 2686976);      // 128 KB
  float*    el       = (float*)(ws + 2818048);      //  16 KB
  float*    er       = (float*)(ws + 2834432);      //  16 KB
  float*    g1       = (float*)(ws + 2850816);      // 128 KB
  float*    g1vol    = (float*)(ws + 2981888);      // 128 KB
  float*    g1n      = (float*)(ws + 3112960);      // 128 KB
  float*    g2       = (float*)(ws + 3244032);      // 128 KB
  float*    nodesOut = (float*)(ws + 3375104);      // 128 KB
  unsigned* bexp     = (unsigned*)(ws + 3506176);   //   4 KB
  unsigned* bgate    = (unsigned*)(ws + 3510272);   //   8 KB

  k_pool    <<<8192, 256, 0, stream>>>(x, xp);
  k_compress<<<4,    256, 0, stream>>>(xp, compress_w, compress_b, nodes, nrm16);
  k_sims    <<<512,  128, 0, stream>>>(nrm16, S);
  k_topk    <<<128,  256, 0, stream>>>(S, A);
  k_sym     <<<4,    256, 0, stream>>>(A, adj);
  k_gat_pre <<<4,    256, 0, stream>>>(nodes, gat1_W, gat1_al, gat1_ar, Hb, el, er);
  k_gat_attn<<<128,  256, 0, stream>>>(Hb, el, er, adj, g1);
  k_inorm   <<<8,    256, 0, stream>>>(g1, g1vol, nullptr, g1n);
  k_gat_pre <<<4,    256, 0, stream>>>(g1n, gat2_W, gat2_al, gat2_ar, Hb, el, er);
  k_gat_attn<<<128,  256, 0, stream>>>(Hb, el, er, adj, g2);
  k_inorm   <<<8,    256, 0, stream>>>(g2, nodesOut, g1vol, nullptr);
  k_bfrag   <<<4,    256, 0, stream>>>(expand_w, 64, 32, 1, bexp);
  k_bfrag   <<<8,    256, 0, stream>>>(gate_w,   64, 64, 2, bgate);
  k_fused_out<<<8192,128, 0, stream>>>(x, nodesOut, bexp, bgate, expand_b, gate_b, out);
}
